// VEncoder_60112362275073
// MI455X (gfx1250) — matrix operand order repaired and checked by reading, not compiled, not run on hardware
//
#include <hip/hip_runtime.h>
#include <hip/hip_bf16.h>

typedef __bf16 bf16_t;
typedef __attribute__((ext_vector_type(16))) __bf16 v16bf;
typedef __attribute__((ext_vector_type(8)))  __bf16 v8bf;
typedef __attribute__((ext_vector_type(8)))  float  v8f;
typedef __attribute__((ext_vector_type(4)))  int    v4i;

#define NB   16
#define NT   512
#define ND   512
#define NAD  128
#define NH   8
#define NDK  64
#define NDV  64
#define NDI  2048
#define NL   6
#define NROWS (NB*NT)

#if defined(__has_builtin)
#if __has_builtin(__builtin_amdgcn_global_load_async_to_lds_b128)
#define HAVE_ASYNC_LDS 1
#endif
#endif

#define AS_GLOBAL __attribute__((address_space(1)))
#define AS_LDS    __attribute__((address_space(3)))

// ---------------------------------------------------------------- helpers
__device__ inline float block_reduce_sum(float v, float* sh) {
#pragma unroll
  for (int off = 16; off > 0; off >>= 1) v += __shfl_xor(v, off, 32);
  const int lane = threadIdx.x & 31, wid = threadIdx.x >> 5;
  const int nw = blockDim.x >> 5;
  if (lane == 0) sh[wid] = v;
  __syncthreads();
  float tot = 0.f;
  for (int i = 0; i < nw; ++i) tot += sh[i];
  __syncthreads();
  return tot;
}

// ---------------------------------------------------------------- f32 -> bf16
__global__ void f32_to_bf16_kernel(const float* __restrict__ s, bf16_t* __restrict__ d, int n) {
  int i = blockIdx.x * 256 + threadIdx.x;
  if (i < n) d[i] = (bf16_t)s[i];
}

// ---------------------------------------------------------------- positional table
__global__ void pos_table_kernel(float* __restrict__ P) {
  int i = blockIdx.x * 256 + threadIdx.x;           // T*D elements
  int t = i / ND, j = i % ND;
  float ex = -(float)(2 * (j / 2)) / (float)ND;
  float angle = (float)t * __expf(ex * __logf(10000.0f));
  P[i] = (j & 1) ? __cosf(angle) : __sinf(angle);
}

// ---------------------------------------------------------------- generic WMMA GEMM
// C[M,N] = A[M,K] @ W[N,K]^T (+bias) (+res) (relu?) -> fp32 Cf and/or bf16 Cb
// Block: 128 threads = 4 waves; wave computes 32x64; block computes 128x64.
// B tile (64 cols x 32 K, bf16) double-buffered in LDS via async global->LDS.
// Requires M % 128 == 0, K % 32 == 0, N % 64 == 0 (true for all call sites).
__global__ void __launch_bounds__(128) gemm_bf16_wmma_kernel(
    const bf16_t* __restrict__ A, int lda,
    const bf16_t* __restrict__ W, int ldw,
    const float* __restrict__ bias,
    const float* __restrict__ res, int ldr,
    float* __restrict__ Cf, int ldc,
    bf16_t* __restrict__ Cb, int ldcb,
    int M, int N, int K, int relu)
{
  const int tid  = threadIdx.x;
  const int lane = tid & 31;
  const int wv   = tid >> 5;
  const int hf   = lane >> 4;        // which 16-lane half
  const int r    = lane & 15;
  const int m0 = (blockIdx.y * 4 + wv) * 32;
  const int n0 = blockIdx.x * 64;

  __shared__ bf16_t Bs[2][64][32];   // [buf][n within tile][k within tile], 8KB

  // cooperative B-tile stage: 64 rows x 64B; 256 16B-chunks; 2 per thread
  auto stage = [&](int kk, int buf) {
#pragma unroll
    for (int c = 0; c < 2; ++c) {
      const int chunk = tid * 2 + c;
      const int row = chunk >> 2;          // 0..63
      const int seg = chunk & 3;           // 0..3 (16B segments)
      const bf16_t* gp = W + (size_t)(n0 + row) * ldw + kk + seg * 8;
      bf16_t* lp = &Bs[buf][row][seg * 8];
#if HAVE_ASYNC_LDS
      __builtin_amdgcn_global_load_async_to_lds_b128(
          (AS_GLOBAL v4i*)gp, (AS_LDS v4i*)lp, 0, 0);
#else
      *(v8bf*)lp = *(const v8bf*)gp;
#endif
    }
  };

  v8f acc[2][4];
#pragma unroll
  for (int g = 0; g < 2; ++g)
#pragma unroll
    for (int t = 0; t < 4; ++t) acc[g][t] = (v8f){};

  const bf16_t* aRow0 = A + (size_t)(m0 + r) * lda;
  const bf16_t* aRow1 = A + (size_t)(m0 + 16 + r) * lda;

  const int S = K >> 5;                // 32-wide K stages
  stage(0, 0);

  for (int s = 0; s < S; ++s) {
    const int k = s << 5;
    if (s + 1 < S) {
      stage(k + 32, (s + 1) & 1);
#if HAVE_ASYNC_LDS
      asm volatile("s_wait_asynccnt 2" ::: "memory");  // in-order: stage s landed
#endif
    } else {
#if HAVE_ASYNC_LDS
      asm volatile("s_wait_asynccnt 0" ::: "memory");
#endif
    }
    __syncthreads();                   // B tile for stage s visible to all waves
    const int buf = s & 1;

    // A fragments for two 16-row groups (lane=M row; halves split K)
    v16bf af0, af1;
    {
      v8bf a0l = *(const v8bf*)(aRow0 + k + hf * 16);
      v8bf a0h = *(const v8bf*)(aRow0 + k + hf * 16 + 8);
      v8bf a1l = *(const v8bf*)(aRow1 + k + hf * 16);
      v8bf a1h = *(const v8bf*)(aRow1 + k + hf * 16 + 8);
#pragma unroll
      for (int i = 0; i < 8; ++i) {
        af0[i] = a0l[i]; af0[i + 8] = a0h[i];
        af1[i] = a1l[i]; af1[i + 8] = a1h[i];
      }
    }
    __builtin_prefetch(aRow0 + k + 64, 0, 3);
    __builtin_prefetch(aRow1 + k + 64, 0, 3);

    // B fragments from LDS: lane=N row; 16 contiguous K per half
    v16bf b0 = *(const v16bf*)&Bs[buf][r][hf * 16];
    v16bf b1 = *(const v16bf*)&Bs[buf][16 + r][hf * 16];
    v16bf b2 = *(const v16bf*)&Bs[buf][32 + r][hf * 16];
    v16bf b3 = *(const v16bf*)&Bs[buf][48 + r][hf * 16];

    acc[0][0] = __builtin_amdgcn_wmma_f32_16x16x32_bf16(false, af0, false, b0, (short)0, acc[0][0], false, false);
    acc[0][1] = __builtin_amdgcn_wmma_f32_16x16x32_bf16(false, af0, false, b1, (short)0, acc[0][1], false, false);
    acc[0][2] = __builtin_amdgcn_wmma_f32_16x16x32_bf16(false, af0, false, b2, (short)0, acc[0][2], false, false);
    acc[0][3] = __builtin_amdgcn_wmma_f32_16x16x32_bf16(false, af0, false, b3, (short)0, acc[0][3], false, false);
    acc[1][0] = __builtin_amdgcn_wmma_f32_16x16x32_bf16(false, af1, false, b0, (short)0, acc[1][0], false, false);
    acc[1][1] = __builtin_amdgcn_wmma_f32_16x16x32_bf16(false, af1, false, b1, (short)0, acc[1][1], false, false);
    acc[1][2] = __builtin_amdgcn_wmma_f32_16x16x32_bf16(false, af1, false, b2, (short)0, acc[1][2], false, false);
    acc[1][3] = __builtin_amdgcn_wmma_f32_16x16x32_bf16(false, af1, false, b3, (short)0, acc[1][3], false, false);

    __syncthreads();                   // all waves done reading buf before restage
  }

#pragma unroll
  for (int g = 0; g < 2; ++g) {
#pragma unroll
    for (int t = 0; t < 4; ++t) {
      const int col = n0 + t * 16 + r;
      const float bv = bias ? bias[col] : 0.f;
#pragma unroll
      for (int i = 0; i < 8; ++i) {
        const int row = m0 + g * 16 + hf * 8 + i;
        float v = acc[g][t][i] + bv;
        if (res)  v += res[(size_t)row * ldr + col];
        if (relu) v = v > 0.f ? v : 0.f;
        if (Cf) Cf[(size_t)row * ldc  + col] = v;
        if (Cb) Cb[(size_t)row * ldcb + col] = (bf16_t)v;
      }
    }
  }
}

// ---------------------------------------------------------------- flash attention
// Q,K: bf16 [B*T, H*DK]; Vt: bf16 [B*H*DV, T]; O: bf16 [B*T, H*DV]
__global__ void __launch_bounds__(128) attention_flash_kernel(
    const bf16_t* __restrict__ Q, const bf16_t* __restrict__ Kmat,
    const bf16_t* __restrict__ Vt, const int* __restrict__ mask,
    bf16_t* __restrict__ O, float scale)
{
  const int lane = threadIdx.x & 31;
  const int wv   = threadIdx.x >> 5;
  const int hf   = lane >> 4;
  const int r    = lane & 15;
  const int bh   = blockIdx.x;
  const int b    = bh >> 3;
  const int head = bh & 7;
  const int r0   = blockIdx.y * 64 + wv * 16;

  __shared__ bf16_t pbuf[4][16][32];

  // Q fragments (two 32-wide dk chunks), resident for the whole loop
  const bf16_t* qrow = Q + (size_t)(b * NT + r0 + r) * (NH * NDK) + head * NDK;
  v16bf qa0, qa1;
  {
    v8bf l0 = *(const v8bf*)(qrow + hf * 16);
    v8bf h0 = *(const v8bf*)(qrow + 8 + hf * 16);
    v8bf l1 = *(const v8bf*)(qrow + 32 + hf * 16);
    v8bf h1 = *(const v8bf*)(qrow + 40 + hf * 16);
#pragma unroll
    for (int i = 0; i < 8; ++i) { qa0[i] = l0[i]; qa0[i + 8] = h0[i]; qa1[i] = l1[i]; qa1[i + 8] = h1[i]; }
  }

  float m_run[8], l_run[8];
#pragma unroll
  for (int i = 0; i < 8; ++i) { m_run[i] = -1e30f; l_run[i] = 0.f; }
  v8f o0 = {}, o1 = {}, o2 = {}, o3 = {};

  const bf16_t* vbase = Vt + ((size_t)bh * NDV + r) * NT;

  for (int kt = 0; kt < NT; kt += 32) {
    // ---- scores: two 16-col tiles, K-dim = dk (64 -> two chunks)
    v8f s0 = {}, s1 = {};
    {
      const bf16_t* kr0 = Kmat + (size_t)(b * NT + kt + r) * (NH * NDK) + head * NDK;
      v16bf kb0 = *(const v16bf*)(kr0 + hf * 16);
      v16bf kb1 = *(const v16bf*)(kr0 + 32 + hf * 16);
      s0 = __builtin_amdgcn_wmma_f32_16x16x32_bf16(false, qa0, false, kb0, (short)0, s0, false, false);
      s0 = __builtin_amdgcn_wmma_f32_16x16x32_bf16(false, qa1, false, kb1, (short)0, s0, false, false);
      const bf16_t* kr1 = kr0 + (size_t)16 * (NH * NDK);
      v16bf kc0 = *(const v16bf*)(kr1 + hf * 16);
      v16bf kc1 = *(const v16bf*)(kr1 + 32 + hf * 16);
      s1 = __builtin_amdgcn_wmma_f32_16x16x32_bf16(false, qa0, false, kc0, (short)0, s1, false, false);
      s1 = __builtin_amdgcn_wmma_f32_16x16x32_bf16(false, qa1, false, kc1, (short)0, s1, false, false);
    }
    const bool dead0 = (mask[b * NT + kt + r] == 0);
    const bool dead1 = (mask[b * NT + kt + 16 + r] == 0);

    // ---- online softmax (rows = hf*8+i; 16-lane reductions stay within halves)
#pragma unroll
    for (int i = 0; i < 8; ++i) {
      float a0 = dead0 ? -1e9f : s0[i] * scale;
      float a1 = dead1 ? -1e9f : s1[i] * scale;
      float tm = fmaxf(a0, a1);
#pragma unroll
      for (int off = 1; off < 16; off <<= 1) tm = fmaxf(tm, __shfl_xor(tm, off, 32));
      float mn = fmaxf(m_run[i], tm);
      float f  = __expf(m_run[i] - mn);
      float e0 = __expf(a0 - mn);
      float e1 = __expf(a1 - mn);
      float es = e0 + e1;
#pragma unroll
      for (int off = 1; off < 16; off <<= 1) es += __shfl_xor(es, off, 32);
      l_run[i] = l_run[i] * f + es;
      m_run[i] = mn;
      o0[i] *= f; o1[i] *= f; o2[i] *= f; o3[i] *= f;
      pbuf[wv][hf * 8 + i][r]      = (bf16_t)e0;
      pbuf[wv][hf * 8 + i][16 + r] = (bf16_t)e1;
    }
    asm volatile("s_wait_dscnt 0" ::: "memory");  // wave-local C-layout -> A-layout via LDS

    v16bf pa;
    {
      v8bf plo = *(const v8bf*)&pbuf[wv][r][hf * 16];
      v8bf phi = *(const v8bf*)&pbuf[wv][r][hf * 16 + 8];
#pragma unroll
      for (int i = 0; i < 8; ++i) { pa[i] = plo[i]; pa[i + 8] = phi[i]; }
    }

    // ---- P @ V : K-dim = 32 key positions, N = dv (4 tiles)
    const bf16_t* vp = vbase + kt + hf * 16;
    v16bf vb0 = *(const v16bf*)(vp);
    v16bf vb1 = *(const v16bf*)(vp + (size_t)16 * NT);
    v16bf vb2 = *(const v16bf*)(vp + (size_t)32 * NT);
    v16bf vb3 = *(const v16bf*)(vp + (size_t)48 * NT);
    o0 = __builtin_amdgcn_wmma_f32_16x16x32_bf16(false, pa, false, vb0, (short)0, o0, false, false);
    o1 = __builtin_amdgcn_wmma_f32_16x16x32_bf16(false, pa, false, vb1, (short)0, o1, false, false);
    o2 = __builtin_amdgcn_wmma_f32_16x16x32_bf16(false, pa, false, vb2, (short)0, o2, false, false);
    o3 = __builtin_amdgcn_wmma_f32_16x16x32_bf16(false, pa, false, vb3, (short)0, o3, false, false);
  }

  v8f oo[4] = {o0, o1, o2, o3};
#pragma unroll
  for (int t = 0; t < 4; ++t) {
#pragma unroll
    for (int i = 0; i < 8; ++i) {
      const int grow = b * NT + r0 + hf * 8 + i;
      const int col  = head * NDV + t * 16 + r;
      O[(size_t)grow * (NH * NDV) + col] = (bf16_t)(oo[t][i] / l_run[i]);
    }
  }
}

// ---------------------------------------------------------------- layernorm (row of 512)
__global__ void __launch_bounds__(256) layernorm_kernel(
    const float* __restrict__ X, const float* __restrict__ Res,
    const float* __restrict__ Pos,
    const float* __restrict__ g, const float* __restrict__ bb,
    float* __restrict__ Yf, bf16_t* __restrict__ Yb, float eps)
{
  __shared__ float sh[8];
  const int row = blockIdx.x;
  const int j0 = threadIdx.x, j1 = threadIdx.x + 256;
  const size_t base = (size_t)row * ND;
  float v0 = X[base + j0], v1 = X[base + j1];
  if (Res) { v0 += Res[base + j0]; v1 += Res[base + j1]; }
  float s  = block_reduce_sum(v0 + v1, sh);
  float sq = block_reduce_sum(v0 * v0 + v1 * v1, sh);
  float mean = s * (1.f / ND);
  float var  = sq * (1.f / ND) - mean * mean;
  float rs   = rsqrtf(var + eps);
  const int t = row & (NT - 1);
  float y0 = (v0 - mean) * rs * g[j0] + bb[j0];
  float y1 = (v1 - mean) * rs * g[j1] + bb[j1];
  if (Pos) { y0 += Pos[(size_t)t * ND + j0]; y1 += Pos[(size_t)t * ND + j1]; }
  if (Yf) { Yf[base + j0] = y0; Yf[base + j1] = y1; }
  if (Yb) { Yb[base + j0] = (bf16_t)y0; Yb[base + j1] = (bf16_t)y1; }
}

// ---------------------------------------------------------------- graph constructor
__global__ void gc_nodevec_kernel(const float* __restrict__ emb, const float* __restrict__ W,
                                  const float* __restrict__ bias, float* __restrict__ out) {
  const int v = blockIdx.x, j = threadIdx.x;      // 512 blocks x 64 threads
  float s = bias[j];
  for (int k = 0; k < NDK; ++k) s += emb[v * NDK + k] * W[j * NDK + k];
  out[v * NDK + j] = tanhf(3.0f * s);
}

__global__ void gc_adj_kernel(const float* __restrict__ n1, const float* __restrict__ n2,
                              float* __restrict__ adj) {
  const int idx = blockIdx.x * 256 + threadIdx.x;  // 512*512
  const int v = idx >> 9, w = idx & 511;
  float sa = 0.f, sb = 0.f;
  for (int k = 0; k < NDK; ++k) {
    sa += n1[v * NDK + k] * n2[w * NDK + k];
    sb += n2[v * NDK + k] * n1[w * NDK + k];
  }
  adj[idx] = fmaxf(0.f, tanhf(3.0f * (sa - sb)));
}

__global__ void __launch_bounds__(256) gc_top2_kernel(float* __restrict__ adj) {
  const int v = blockIdx.x;
  float* row = adj + (size_t)v * ND;
  __shared__ float sv[256];
  __shared__ int   si[256];
  float b1v = -1e30f; int b1i = 0;
  for (int w = threadIdx.x; w < ND; w += 256)
    if (row[w] > b1v) { b1v = row[w]; b1i = w; }
  sv[threadIdx.x] = b1v; si[threadIdx.x] = b1i;
  __syncthreads();
  for (int st = 128; st > 0; st >>= 1) {
    if (threadIdx.x < st && sv[threadIdx.x + st] > sv[threadIdx.x]) {
      sv[threadIdx.x] = sv[threadIdx.x + st]; si[threadIdx.x] = si[threadIdx.x + st];
    }
    __syncthreads();
  }
  const int i1 = si[0];
  __syncthreads();
  float b2v = -1e30f; int b2i = 0;
  for (int w = threadIdx.x; w < ND; w += 256)
    if (w != i1 && row[w] > b2v) { b2v = row[w]; b2i = w; }
  sv[threadIdx.x] = b2v; si[threadIdx.x] = b2i;
  __syncthreads();
  for (int st = 128; st > 0; st >>= 1) {
    if (threadIdx.x < st && sv[threadIdx.x + st] > sv[threadIdx.x]) {
      sv[threadIdx.x] = sv[threadIdx.x + st]; si[threadIdx.x] = si[threadIdx.x + st];
    }
    __syncthreads();
  }
  const int i2 = si[0];
  __syncthreads();
  for (int w = threadIdx.x; w < ND; w += 256)
    if (w != i1 && w != i2) row[w] = 0.f;
}

// Build transposed row-normalized adjacencies (bf16, [w,v] layout for GEMM W[N,K])
__global__ void __launch_bounds__(256) gc_an_kernel(const float* __restrict__ adj,
                                                    bf16_t* __restrict__ AnLT,
                                                    bf16_t* __restrict__ AnRT) {
  __shared__ float sh[8];
  const int v = blockIdx.x;
  float sl = 0.f, sr = 0.f;
  for (int w = threadIdx.x; w < ND; w += 256) {
    sl += adj[(size_t)v * ND + w];
    sr += adj[(size_t)w * ND + v];
  }
  sl = block_reduce_sum(sl, sh);
  sr = block_reduce_sum(sr, sh);
  const float rl = 1.f / (1.f + sl);
  const float rr = 1.f / (1.f + sr);
  for (int w = threadIdx.x; w < ND; w += 256) {
    float eye = (v == w) ? 1.f : 0.f;
    AnLT[(size_t)w * ND + v] = (bf16_t)((adj[(size_t)v * ND + w] + eye) * rl);
    AnRT[(size_t)w * ND + v] = (bf16_t)((adj[(size_t)w * ND + v] + eye) * rr);
  }
}

// h_next = 0.2*x + 0.8*G (or just x if G==null), written to concat slot (bf16, ld 1536)
__global__ void mix_combine_kernel(const float* __restrict__ x, const float* __restrict__ G,
                                   bf16_t* __restrict__ cc, int slot) {
  const int i = blockIdx.x * 256 + threadIdx.x;     // NROWS*ND
  const int row = i >> 9, w = i & 511;
  float val = G ? (0.2f * x[i] + 0.8f * G[i]) : x[i];
  cc[(size_t)row * (3 * ND) + slot * ND + w] = (bf16_t)val;
}

// vbf [B*T, H*DV] -> vt [B*H*DV, T]
__global__ void v_transpose_kernel(const bf16_t* __restrict__ vbf, bf16_t* __restrict__ vt) {
  const int i = blockIdx.x * 256 + threadIdx.x;     // NROWS*ND
  const int row = i >> 9, col = i & 511;
  const int b = row >> 9, t = row & 511;            // T == 512
  const int h = col >> 6, dv = col & 63;
  vt[(((size_t)(b * NH + h) * NDV) + dv) * NT + t] = vbf[i];
}

// BatchNorm1d over channel T: stats over (batch, feature)
__global__ void __launch_bounds__(256) batchnorm_kernel(const float* __restrict__ mu,
                                                        const float* __restrict__ bnb,
                                                        float* __restrict__ out) {
  __shared__ float sh[8];
  const int t = blockIdx.x;
  float s = 0.f, sq = 0.f;
  for (int i = threadIdx.x; i < NB * ND; i += 256) {
    int b = i >> 9, d = i & 511;
    float v = mu[((size_t)b * NT + t) * ND + d];
    s += v; sq += v * v;
  }
  s  = block_reduce_sum(s, sh);
  sq = block_reduce_sum(sq, sh);
  const float inv = 1.f / (float)(NB * ND);
  float mean = s * inv;
  float var  = sq * inv - mean * mean;
  float rs   = rsqrtf(var + 1e-5f);
  float beta = bnb[t];
  for (int i = threadIdx.x; i < NB * ND; i += 256) {
    int b = i >> 9, d = i & 511;
    size_t idx = ((size_t)b * NT + t) * ND + d;
    out[idx] = (mu[idx] - mean) * rs + beta;
  }
}

// ================================================================= launch
extern "C" void kernel_launch(void* const* d_in, const int* in_sizes, int n_in,
                              void* d_out, int out_size, void* d_ws, size_t ws_size,
                              hipStream_t stream) {
  (void)in_sizes; (void)n_in; (void)out_size; (void)ws_size;
  const float* src_seq   = (const float*)d_in[0];
  const int*   src_mask  = (const int*)d_in[1];
  const float* emb_W     = (const float*)d_in[2];
  const float* emb_b     = (const float*)d_in[3];
  const float* gc_emb1   = (const float*)d_in[4];
  const float* gc_emb2   = (const float*)d_in[5];
  const float* gc_lin1_W = (const float*)d_in[6];
  const float* gc_lin1_b = (const float*)d_in[7];
  const float* gc_lin2_W = (const float*)d_in[8];
  const float* gc_lin2_b = (const float*)d_in[9];
  const float* mixL_W    = (const float*)d_in[10];
  const float* mixL_b    = (const float*)d_in[11];
  const float* mixR_W    = (const float*)d_in[12];
  const float* mixR_b    = (const float*)d_in[13];
  const float* gcn_ln_g  = (const float*)d_in[14];
  const float* gcn_ln_b  = (const float*)d_in[15];
  const float* Wq        = (const float*)d_in[16];
  const float* Wk        = (const float*)d_in[17];
  const float* Wv        = (const float*)d_in[18];
  const float* Wo        = (const float*)d_in[19];
  const float* ln1_g     = (const float*)d_in[20];
  const float* ln1_b     = (const float*)d_in[21];
  const float* W1        = (const float*)d_in[22];
  const float* b1        = (const float*)d_in[23];
  const float* W2        = (const float*)d_in[24];
  const float* b2        = (const float*)d_in[25];
  const float* ln2_g     = (const float*)d_in[26];
  const float* ln2_b     = (const float*)d_in[27];
  const float* fln_g     = (const float*)d_in[28];
  const float* fln_b     = (const float*)d_in[29];
  const float* mu_W      = (const float*)d_in[30];
  const float* mu_b      = (const float*)d_in[31];
  const float* bn_b      = (const float*)d_in[32];
  const float* lv_W      = (const float*)d_in[33];
  const float* lv_b      = (const float*)d_in[34];

  char* ws = (char*)d_ws;
  size_t off = 0;
  auto take = [&](size_t elems, size_t esz) -> char* {
    char* p = ws + off;
    off += (elems * esz + 255) & ~(size_t)255;
    return p;
  };

  bf16_t* srcBf = (bf16_t*)take((size_t)NB * NT * NAD, 2);
  bf16_t* embWb = (bf16_t*)take((size_t)ND * NAD, 2);
  bf16_t* mixLb = (bf16_t*)take((size_t)ND * 3 * ND, 2);
  bf16_t* mixRb = (bf16_t*)take((size_t)ND * 3 * ND, 2);
  bf16_t* WqB   = (bf16_t*)take((size_t)NL * ND * ND, 2);
  bf16_t* WkB   = (bf16_t*)take((size_t)NL * ND * ND, 2);
  bf16_t* WvB   = (bf16_t*)take((size_t)NL * ND * ND, 2);
  bf16_t* WoB   = (bf16_t*)take((size_t)NL * ND * ND, 2);
  bf16_t* W1B   = (bf16_t*)take((size_t)NL * NDI * ND, 2);
  bf16_t* W2B   = (bf16_t*)take((size_t)NL * ND * NDI, 2);
  bf16_t* muWB  = (bf16_t*)take((size_t)ND * ND, 2);
  bf16_t* lvWB  = (bf16_t*)take((size_t)ND * ND, 2);
  bf16_t* AnLT  = (bf16_t*)take((size_t)ND * ND, 2);
  bf16_t* AnRT  = (bf16_t*)take((size_t)ND * ND, 2);
  float*  posT  = (float*)take((size_t)NT * ND, 4);
  float*  n1b   = (float*)take((size_t)ND * NDK, 4);
  float*  n2b   = (float*)take((size_t)ND * NDK, 4);
  float*  adj   = (float*)take((size_t)ND * ND, 4);
  float*  x0    = (float*)take((size_t)NROWS * ND, 4);
  float*  enc   = (float*)take((size_t)NROWS * ND, 4);
  bf16_t* hbf   = (bf16_t*)take((size_t)NROWS * ND, 2);
  float*  Gtmp  = (float*)take((size_t)NROWS * ND, 4);   // reused as mu buffer at the end
  float*  ho    = (float*)take((size_t)NROWS * ND, 4);

  // union region: (ccL,ccR) during GCN phase; (q,k,v,vt,o,f) during layer phase
  char* U = ws + off;
  bf16_t* ccL = (bf16_t*)U;
  bf16_t* ccR = ccL + (size_t)NROWS * 3 * ND;
  bf16_t* qbf = (bf16_t*)U;
  bf16_t* kbf = qbf + (size_t)NROWS * ND;
  bf16_t* vbf = kbf + (size_t)NROWS * ND;
  bf16_t* vtb = vbf + (size_t)NROWS * ND;
  bf16_t* obf = vtb + (size_t)NROWS * ND;
  bf16_t* fbf = obf + (size_t)NROWS * ND;                // NROWS*NDI

  auto cvt = [&](const float* s, bf16_t* d, size_t n) {
    f32_to_bf16_kernel<<<(unsigned)((n + 255) / 256), 256, 0, stream>>>(s, d, (int)n);
  };
  auto gemm = [&](const bf16_t* A, int lda, const bf16_t* W_, int ldw, const float* bias_,
                  const float* res, int ldr, float* Cf, int ldc, bf16_t* Cb, int ldcb,
                  int M, int N, int K, int relu) {
    dim3 grid(N / 64, M / 128), blk(128);
    gemm_bf16_wmma_kernel<<<grid, blk, 0, stream>>>(A, lda, W_, ldw, bias_, res, ldr,
                                                    Cf, ldc, Cb, ldcb, M, N, K, relu);
  };

  // ---- weight / input conversions (weights stay hot in 192MB L2)
  cvt(src_seq, srcBf, (size_t)NB * NT * NAD);
  cvt(emb_W, embWb, (size_t)ND * NAD);
  cvt(mixL_W, mixLb, (size_t)ND * 3 * ND);
  cvt(mixR_W, mixRb, (size_t)ND * 3 * ND);
  cvt(Wq, WqB, (size_t)NL * ND * ND);
  cvt(Wk, WkB, (size_t)NL * ND * ND);
  cvt(Wv, WvB, (size_t)NL * ND * ND);
  cvt(Wo, WoB, (size_t)NL * ND * ND);
  cvt(W1, W1B, (size_t)NL * NDI * ND);
  cvt(W2, W2B, (size_t)NL * ND * NDI);
  cvt(mu_W, muWB, (size_t)ND * ND);
  cvt(lv_W, lvWB, (size_t)ND * ND);

  pos_table_kernel<<<NT * ND / 256, 256, 0, stream>>>(posT);

  // ---- graph constructor (tiny)
  gc_nodevec_kernel<<<ND, 64, 0, stream>>>(gc_emb1, gc_lin1_W, gc_lin1_b, n1b);
  gc_nodevec_kernel<<<ND, 64, 0, stream>>>(gc_emb2, gc_lin2_W, gc_lin2_b, n2b);
  gc_adj_kernel<<<ND * ND / 256, 256, 0, stream>>>(n1b, n2b, adj);
  gc_top2_kernel<<<ND, 256, 0, stream>>>(adj);
  gc_an_kernel<<<ND, 256, 0, stream>>>(adj, AnLT, AnRT);

  // ---- embedding: x0 fp32 + bf16 into concat slot 0
  gemm(srcBf, NAD, embWb, NAD, emb_b, nullptr, 0, x0, ND, ccL, 3 * ND, NROWS, ND, NAD, 0);
  mix_combine_kernel<<<NROWS * ND / 256, 256, 0, stream>>>(x0, nullptr, ccR, 0);

  // ---- mixprop L (An) and R (An of adj^T)
  gemm(ccL, 3 * ND, AnLT, ND, nullptr, nullptr, 0, Gtmp, ND, nullptr, 0, NROWS, ND, ND, 0);
  mix_combine_kernel<<<NROWS * ND / 256, 256, 0, stream>>>(x0, Gtmp, ccL, 1);
  gemm(ccL + ND, 3 * ND, AnLT, ND, nullptr, nullptr, 0, Gtmp, ND, nullptr, 0, NROWS, ND, ND, 0);
  mix_combine_kernel<<<NROWS * ND / 256, 256, 0, stream>>>(x0, Gtmp, ccL, 2);
  gemm(ccR, 3 * ND, AnRT, ND, nullptr, nullptr, 0, Gtmp, ND, nullptr, 0, NROWS, ND, ND, 0);
  mix_combine_kernel<<<NROWS * ND / 256, 256, 0, stream>>>(x0, Gtmp, ccR, 1);
  gemm(ccR + ND, 3 * ND, AnRT, ND, nullptr, nullptr, 0, Gtmp, ND, nullptr, 0, NROWS, ND, ND, 0);
  mix_combine_kernel<<<NROWS * ND / 256, 256, 0, stream>>>(x0, Gtmp, ccR, 2);

  gemm(ccL, 3 * ND, mixLb, 3 * ND, mixL_b, nullptr, 0, ho, ND, nullptr, 0, NROWS, ND, 3 * ND, 0);
  gemm(ccR, 3 * ND, mixRb, 3 * ND, mixR_b, ho, ND, ho, ND, nullptr, 0, NROWS, ND, 3 * ND, 0);

  // x = LN(ho + x0); enc = x + pos
  layernorm_kernel<<<NROWS, 256, 0, stream>>>(ho, x0, posT, gcn_ln_g, gcn_ln_b, enc, nullptr, 1e-6f);

  // ---- transformer layers
  const float scale = 0.125f;  // 1/sqrt(64)
  for (int l = 0; l < NL; ++l) {
    layernorm_kernel<<<NROWS, 256, 0, stream>>>(enc, nullptr, nullptr,
        ln1_g + l * ND, ln1_b + l * ND, nullptr, hbf, 1e-6f);
    gemm(hbf, ND, WqB + (size_t)l * ND * ND, ND, nullptr, nullptr, 0, nullptr, 0, qbf, ND, NROWS, ND, ND, 0);
    gemm(hbf, ND, WkB + (size_t)l * ND * ND, ND, nullptr, nullptr, 0, nullptr, 0, kbf, ND, NROWS, ND, ND, 0);
    gemm(hbf, ND, WvB + (size_t)l * ND * ND, ND, nullptr, nullptr, 0, nullptr, 0, vbf, ND, NROWS, ND, ND, 0);
    v_transpose_kernel<<<NROWS * ND / 256, 256, 0, stream>>>(vbf, vtb);
    attention_flash_kernel<<<dim3(NB * NH, NT / 64), 128, 0, stream>>>(qbf, kbf, vtb, src_mask, obf, scale);
    gemm(obf, ND, WoB + (size_t)l * ND * ND, ND, nullptr, enc, ND, enc, ND, nullptr, 0, NROWS, ND, ND, 0);

    layernorm_kernel<<<NROWS, 256, 0, stream>>>(enc, nullptr, nullptr,
        ln2_g + l * ND, ln2_b + l * ND, nullptr, hbf, 1e-6f);
    gemm(hbf, ND, W1B + (size_t)l * NDI * ND, ND, b1 + l * NDI, nullptr, 0,
         nullptr, 0, fbf, NDI, NROWS, NDI, ND, 1);
    gemm(fbf, NDI, W2B + (size_t)l * ND * NDI, NDI, b2 + l * ND, enc, ND,
         enc, ND, nullptr, 0, NROWS, ND, NDI, 0);
  }

  // ---- head
  layernorm_kernel<<<NROWS, 256, 0, stream>>>(enc, nullptr, nullptr, fln_g, fln_b, nullptr, hbf, 1e-6f);
  float* outMean = (float*)d_out;
  float* outLv   = outMean + (size_t)NROWS * ND;
  gemm(hbf, ND, muWB, ND, mu_b, nullptr, 0, Gtmp, ND, nullptr, 0, NROWS, ND, ND, 0);  // mu -> Gtmp
  gemm(hbf, ND, lvWB, ND, lv_b, nullptr, 0, outLv, ND, nullptr, 0, NROWS, ND, ND, 0); // logvar
  batchnorm_kernel<<<NT, 256, 0, stream>>>(Gtmp, bn_b, outMean);
}
